// ColorGNNEmbedding_59287728554041
// MI455X (gfx1250) — matrix operand order repaired and checked by reading, not compiled, not run on hardware
//
#include <hip/hip_runtime.h>
#include <hip/hip_bf16.h>

typedef __attribute__((ext_vector_type(16))) __bf16 v16bf;
typedef __attribute__((ext_vector_type(8)))  __bf16 v8bf;
typedef __attribute__((ext_vector_type(8)))  float  v8f;

#define NNODES 50000
#define NEDGES 800000
#define BN_EPS 1e-5f
#define LRELU_SLOPE 0.01f

// ---------------------------------------------------------------------------
// 1. Build h0 (50000 x 1760 bf16): [layer_emb(250) | resnet(1000) | size_emb(250)
//    | color_emb(3x85=255) | zero pad(5)]
// ---------------------------------------------------------------------------
__global__ __launch_bounds__(256) void embed_kernel(
    const float* __restrict__ x, const float* __restrict__ layer_emb,
    const float* __restrict__ size_emb, const float* __restrict__ color_emb,
    __bf16* __restrict__ h0)
{
    int node = blockIdx.x;
    if (node >= NNODES) return;
    const float* xr = x + (size_t)node * 1005;
    int li = (int)xr[0];
    int si = (int)roundf(xr[1001] * 10.0f);
    si = si < 0 ? 0 : (si > 10 ? 10 : si);
    int ci0 = (int)xr[1002], ci1 = (int)xr[1003], ci2 = (int)xr[1004];
    __bf16* out = h0 + (size_t)node * 1760;
    for (int c = threadIdx.x; c < 1760; c += blockDim.x) {
        float v;
        if (c < 250)        v = layer_emb[li * 250 + c];
        else if (c < 1250)  v = xr[1 + (c - 250)];
        else if (c < 1500)  v = size_emb[si * 250 + (c - 1250)];
        else if (c < 1585)  v = color_emb[ci0 * 85 + (c - 1500)];
        else if (c < 1670)  v = color_emb[ci1 * 85 + (c - 1585)];
        else if (c < 1755)  v = color_emb[ci2 * 85 + (c - 1670)];
        else                v = 0.0f;
        out[c] = (__bf16)v;
    }
}

// ---------------------------------------------------------------------------
// 2. Convert weight (K x N f32, row-major) -> transposed bf16 (N x Kpad), pad=0
// ---------------------------------------------------------------------------
__global__ __launch_bounds__(256) void wt_convert_kernel(
    const float* __restrict__ W, __bf16* __restrict__ WT, int K, int N, int Kpad)
{
    int idx = blockIdx.x * blockDim.x + threadIdx.x;
    int total = N * Kpad;
    if (idx >= total) return;
    int n = idx / Kpad, k = idx % Kpad;
    WT[idx] = (__bf16)((k < K) ? W[(size_t)k * N + n] : 0.0f);
}

// ---------------------------------------------------------------------------
// 3. Degree / inverse-sqrt-degree
// ---------------------------------------------------------------------------
__global__ __launch_bounds__(256) void zero_f32_kernel(float* __restrict__ p, int n)
{
    int i = blockIdx.x * blockDim.x + threadIdx.x;
    if (i < n) p[i] = 0.0f;
}

__global__ __launch_bounds__(256) void deg_kernel(
    const int* __restrict__ dst, const float* __restrict__ ew, float* __restrict__ deg)
{
    int e = blockIdx.x * blockDim.x + threadIdx.x;
    if (e < NEDGES) atomicAdd(&deg[dst[e]], ew[e]);
}

__global__ __launch_bounds__(256) void dis_kernel(
    const float* __restrict__ deg, float* __restrict__ dis)
{
    int i = blockIdx.x * blockDim.x + threadIdx.x;
    if (i < NNODES) dis[i] = rsqrtf(deg[i] + 1.0f);
}

// ---------------------------------------------------------------------------
// 4. WMMA bf16 GEMM: C(MxN,f32) = A(MxK,bf16 row-major) * BT(NxK,bf16)^T
//    One wave computes a 16x64 output strip (4 accumulators, A reused 4x).
//    Requires M%16==0, N%64==0, K%32==0 (guaranteed by padding).
// ---------------------------------------------------------------------------
__global__ __launch_bounds__(256) void gemm_bf16_wmma_kernel(
    const __bf16* __restrict__ A, const __bf16* __restrict__ BT,
    float* __restrict__ C, int M, int N, int K)
{
    const int lane = threadIdx.x & 31;
    const int wavesPerBlock = blockDim.x >> 5;
    const int wave = blockIdx.x * wavesPerBlock + (threadIdx.x >> 5);
    const int nGroups = N >> 6;
    const int totalWaves = (M >> 4) * nGroups;
    if (wave >= totalWaves) return;          // wave-uniform: EXEC stays all-ones
    const int mT = wave / nGroups;
    const int nG = wave % nGroups;

    const int r = lane & 15;
    const int aOff  = (lane < 16) ? 0 : 16;  // A k order used here: K halves per half-wave (k = 16h + s)
    const int bHalf = (lane < 16) ? 0 : 16;  // B k order used here: K halves per half-wave, same as A
    const __bf16* aPtr  = A + (size_t)(mT * 16 + r) * K + aOff;
    const int colBase = nG * 64 + r;
    const __bf16* bPtr0 = BT + (size_t)(colBase +  0) * K + bHalf;
    const __bf16* bPtr1 = BT + (size_t)(colBase + 16) * K + bHalf;
    const __bf16* bPtr2 = BT + (size_t)(colBase + 32) * K + bHalf;
    const __bf16* bPtr3 = BT + (size_t)(colBase + 48) * K + bHalf;

    v8f c0 = {}, c1 = {}, c2 = {}, c3 = {};
    for (int k0 = 0; k0 < K; k0 += 32) {
        v8bf aLo = *(const v8bf*)(aPtr + k0);
        v8bf aHi = *(const v8bf*)(aPtr + k0 + 8);
        v16bf a;
#pragma unroll
        for (int i = 0; i < 8; ++i) { a[i] = aLo[i]; a[i + 8] = aHi[i]; }
        v16bf b0 = *(const v16bf*)(bPtr0 + k0);
        v16bf b1 = *(const v16bf*)(bPtr1 + k0);
        v16bf b2 = *(const v16bf*)(bPtr2 + k0);
        v16bf b3 = *(const v16bf*)(bPtr3 + k0);
        c0 = __builtin_amdgcn_wmma_f32_16x16x32_bf16(false, a, false, b0, (short)0, c0, false, false);
        c1 = __builtin_amdgcn_wmma_f32_16x16x32_bf16(false, a, false, b1, (short)0, c1, false, false);
        c2 = __builtin_amdgcn_wmma_f32_16x16x32_bf16(false, a, false, b2, (short)0, c2, false, false);
        c3 = __builtin_amdgcn_wmma_f32_16x16x32_bf16(false, a, false, b3, (short)0, c3, false, false);
    }
    // C layout: lanes 0-15 -> rows 0..7 (VGPR i = row i), lanes 16-31 -> rows 8..15
    const int cm = mT * 16 + ((lane < 16) ? 0 : 8);
    float* cRow = C + (size_t)cm * N + nG * 64 + r;
#pragma unroll
    for (int i = 0; i < 8; ++i) {
        float* p = cRow + (size_t)i * N;
        p[0] = c0[i]; p[16] = c1[i]; p[32] = c2[i]; p[48] = c3[i];
    }
}

// ---------------------------------------------------------------------------
// 5. agg init: agg[n,f] = dis[n]^2 * hw[n,f] + bias[f]
// ---------------------------------------------------------------------------
__global__ __launch_bounds__(256) void self_init_kernel(
    const float* __restrict__ hw, const float* __restrict__ dis,
    const float* __restrict__ bias, float* __restrict__ agg, int F)
{
    long long idx = (long long)blockIdx.x * blockDim.x + threadIdx.x;
    long long total = (long long)NNODES * F;
    if (idx >= total) return;
    int n = (int)(idx / F), f = (int)(idx % F);
    float d = dis[n];
    agg[idx] = d * d * hw[idx] + bias[f];
}

// ---------------------------------------------------------------------------
// 6. Edge scatter: agg[dst] += dis[src]*ew*dis[dst] * hw[src]   (one block/edge)
// ---------------------------------------------------------------------------
__global__ void edge_scatter_kernel(
    const int* __restrict__ src, const int* __restrict__ dst,
    const float* __restrict__ ew, const float* __restrict__ dis,
    const float* __restrict__ hw, float* __restrict__ agg, int F)
{
    int e = blockIdx.x;
    int s = src[e], d = dst[e];
    float norm = dis[s] * ew[e] * dis[d];
    const float* hs = hw + (size_t)s * F;
    float* ad = agg + (size_t)d * F;
    for (int f = threadIdx.x; f < F; f += blockDim.x)
        atomicAdd(&ad[f], norm * hs[f]);
}

// ---------------------------------------------------------------------------
// 7. BN stats: one block per column -> meanvar[c]=mean, meanvar[F+c]=var
// ---------------------------------------------------------------------------
__global__ __launch_bounds__(256) void bn_stats_kernel(
    const float* __restrict__ h, float* __restrict__ meanvar, int F)
{
    int c = blockIdx.x;
    int tid = threadIdx.x;
    float s = 0.0f, s2 = 0.0f;
    for (int r = tid; r < NNODES; r += 256) {
        float v = h[(size_t)r * F + c];
        s += v; s2 += v * v;
    }
    __shared__ float sh[256], sh2[256];
    sh[tid] = s; sh2[tid] = s2;
    __syncthreads();
    for (int o = 128; o > 0; o >>= 1) {
        if (tid < o) { sh[tid] += sh[tid + o]; sh2[tid] += sh2[tid + o]; }
        __syncthreads();
    }
    if (tid == 0) {
        float m = sh[0] / (float)NNODES;
        float v = sh2[0] / (float)NNODES - m * m;
        meanvar[c] = m;
        meanvar[F + c] = v;
    }
}

// ---------------------------------------------------------------------------
// 8. BN apply + leaky relu; write bf16 (next GEMM input) or f32 (last layer)
// ---------------------------------------------------------------------------
__global__ __launch_bounds__(256) void bn_apply_kernel(
    const float* __restrict__ agg, const float* __restrict__ meanvar,
    const float* __restrict__ g, const float* __restrict__ be,
    __bf16* __restrict__ out_bf, float* __restrict__ out_f, int F)
{
    long long idx = (long long)blockIdx.x * blockDim.x + threadIdx.x;
    long long total = (long long)NNODES * F;
    if (idx >= total) return;
    int f = (int)(idx % F);
    float m = meanvar[f], v = meanvar[F + f];
    float y = g[f] * (agg[idx] - m) * rsqrtf(v + BN_EPS) + be[f];
    y = (y > 0.0f) ? y : LRELU_SLOPE * y;
    if (out_bf) out_bf[idx] = (__bf16)y;
    else        out_f[idx]  = y;
}

// ---------------------------------------------------------------------------
// 9. Final linear 64 -> 3
// ---------------------------------------------------------------------------
__global__ __launch_bounds__(256) void final_linear_kernel(
    const float* __restrict__ h3, const float* __restrict__ Wo,
    const float* __restrict__ bo, float* __restrict__ out)
{
    int n = blockIdx.x * blockDim.x + threadIdx.x;
    if (n >= NNODES) return;
    float a0 = bo[0], a1 = bo[1], a2 = bo[2];
    const float* hr = h3 + (size_t)n * 64;
#pragma unroll
    for (int f = 0; f < 64; ++f) {
        float v = hr[f];
        a0 += v * Wo[f * 3 + 0];
        a1 += v * Wo[f * 3 + 1];
        a2 += v * Wo[f * 3 + 2];
    }
    out[n * 3 + 0] = a0; out[n * 3 + 1] = a1; out[n * 3 + 2] = a2;
}

// ---------------------------------------------------------------------------
extern "C" void kernel_launch(void* const* d_in, const int* in_sizes, int n_in,
                              void* d_out, int out_size, void* d_ws, size_t ws_size,
                              hipStream_t stream)
{
    const float* x         = (const float*)d_in[0];
    const int*   edge_idx  = (const int*)  d_in[1];
    const float* edge_attr = (const float*)d_in[2];
    const float* layer_emb = (const float*)d_in[3];
    const float* size_emb  = (const float*)d_in[4];
    const float* color_emb = (const float*)d_in[5];
    const float* W1 = (const float*)d_in[6];
    const float* b1 = (const float*)d_in[7];
    const float* g1 = (const float*)d_in[8];
    const float* be1= (const float*)d_in[9];
    const float* W2 = (const float*)d_in[10];
    const float* b2 = (const float*)d_in[11];
    const float* g2 = (const float*)d_in[12];
    const float* be2= (const float*)d_in[13];
    const float* W3 = (const float*)d_in[14];
    const float* b3 = (const float*)d_in[15];
    const float* g3 = (const float*)d_in[16];
    const float* be3= (const float*)d_in[17];
    const float* Wo = (const float*)d_in[18];
    const float* bo = (const float*)d_in[19];
    const int* src = edge_idx;
    const int* dst = edge_idx + NEDGES;
    float* out = (float*)d_out;

    // ---- workspace carve-up (peak ~385 MB, buffers reused across layers) ----
    char* base = (char*)d_ws;
    size_t off = 0;
    auto take = [&](size_t bytes) -> void* {
        void* p = base + off;
        off += (bytes + 255) & ~(size_t)255;
        return p;
    };
    __bf16* h0b = (__bf16*)take((size_t)NNODES * 1760 * 2); // reused: h1b/h2b/h3f
    __bf16* W1T = (__bf16*)take((size_t)512 * 1760 * 2);
    __bf16* W2T = (__bf16*)take((size_t)256 * 512 * 2);
    __bf16* W3T = (__bf16*)take((size_t)64 * 256 * 2);
    float*  hw  = (float*) take((size_t)NNODES * 512 * 4);  // GEMM outputs (all layers)
    float*  agg = (float*) take((size_t)NNODES * 512 * 4);  // aggregation (all layers)
    float*  deg = (float*) take((size_t)NNODES * 4);
    float*  dis = (float*) take((size_t)NNODES * 4);
    float*  mv  = (float*) take((size_t)2 * 512 * 4);
    __bf16* h1b = h0b;                 // 50000x512 bf16 fits in h0b region
    __bf16* h2b = h0b;                 // 50000x256 bf16
    float*  h3f = (float*)h0b;         // 50000x64 f32

    // ---- stage 0: embeddings, weight conversion, degrees ----
    embed_kernel<<<NNODES, 256, 0, stream>>>(x, layer_emb, size_emb, color_emb, h0b);
    wt_convert_kernel<<<(512 * 1760 + 255) / 256, 256, 0, stream>>>(W1, W1T, 1755, 512, 1760);
    wt_convert_kernel<<<(256 * 512  + 255) / 256, 256, 0, stream>>>(W2, W2T, 512, 256, 512);
    wt_convert_kernel<<<(64  * 256  + 255) / 256, 256, 0, stream>>>(W3, W3T, 256, 64, 256);
    zero_f32_kernel<<<(NNODES + 255) / 256, 256, 0, stream>>>(deg, NNODES);
    deg_kernel<<<(NEDGES + 255) / 256, 256, 0, stream>>>(dst, edge_attr, deg);
    dis_kernel<<<(NNODES + 255) / 256, 256, 0, stream>>>(deg, dis);

    const int mTiles = NNODES / 16;  // 3125 exactly

    // ---- layer 1: 1760 -> 512 ----
    {
        const int F = 512, K = 1760;
        int waves = mTiles * (F / 64);
        gemm_bf16_wmma_kernel<<<(waves + 7) / 8, 256, 0, stream>>>(h0b, W1T, hw, NNODES, F, K);
        long long total = (long long)NNODES * F;
        self_init_kernel<<<(unsigned)((total + 255) / 256), 256, 0, stream>>>(hw, dis, b1, agg, F);
        edge_scatter_kernel<<<NEDGES, 256, 0, stream>>>(src, dst, edge_attr, dis, hw, agg, F);
        bn_stats_kernel<<<F, 256, 0, stream>>>(agg, mv, F);
        bn_apply_kernel<<<(unsigned)((total + 255) / 256), 256, 0, stream>>>(agg, mv, g1, be1, h1b, nullptr, F);
    }
    // ---- layer 2: 512 -> 256 ----
    {
        const int F = 256, K = 512;
        int waves = mTiles * (F / 64);
        gemm_bf16_wmma_kernel<<<(waves + 7) / 8, 256, 0, stream>>>(h1b, W2T, hw, NNODES, F, K);
        long long total = (long long)NNODES * F;
        self_init_kernel<<<(unsigned)((total + 255) / 256), 256, 0, stream>>>(hw, dis, b2, agg, F);
        edge_scatter_kernel<<<NEDGES, 256, 0, stream>>>(src, dst, edge_attr, dis, hw, agg, F);
        bn_stats_kernel<<<F, 256, 0, stream>>>(agg, mv, F);
        bn_apply_kernel<<<(unsigned)((total + 255) / 256), 256, 0, stream>>>(agg, mv, g2, be2, h2b, nullptr, F);
    }
    // ---- layer 3: 256 -> 64 ----
    {
        const int F = 64, K = 256;
        int waves = mTiles * (F / 64);
        gemm_bf16_wmma_kernel<<<(waves + 7) / 8, 256, 0, stream>>>(h2b, W3T, hw, NNODES, F, K);
        long long total = (long long)NNODES * F;
        self_init_kernel<<<(unsigned)((total + 255) / 256), 256, 0, stream>>>(hw, dis, b3, agg, F);
        edge_scatter_kernel<<<NEDGES, 64, 0, stream>>>(src, dst, edge_attr, dis, hw, agg, F);
        bn_stats_kernel<<<F, 256, 0, stream>>>(agg, mv, F);
        bn_apply_kernel<<<(unsigned)((total + 255) / 256), 256, 0, stream>>>(agg, mv, g3, be3, nullptr, h3f, F);
    }
    // ---- output: 64 -> 3 ----
    final_linear_kernel<<<(NNODES + 255) / 256, 256, 0, stream>>>(h3f, Wo, bo, out);
}
